// MyCustomAttentionMap_79809082295131
// MI455X (gfx1250) — matrix operand order repaired and checked by reading, not compiled, not run on hardware
//
#include <hip/hip_runtime.h>

#define DEVINL __device__ __forceinline__

typedef __attribute__((ext_vector_type(16))) __bf16        bf16x16;
typedef __attribute__((ext_vector_type(8)))  float         f32x8;
typedef __attribute__((ext_vector_type(4)))  float         f32x4;
typedef __attribute__((ext_vector_type(2)))  float         f32x2;
typedef __attribute__((ext_vector_type(4)))  unsigned int  u32x4;

union AFrag { u32x4 u[2]; bf16x16 v; };

constexpr int   cB = 8, cN = 1024, cC = 768, cH = 12, cD = 64;
constexpr float cSCALE = 0.125f;     // D^-0.5
constexpr int   KT   = 32;           // K step per WMMA
constexpr int   LSTR = KT + 8;       // padded LDS row stride (elements)

// ---------------------------------------------------------------------------
// Async global -> LDS tile copy (bf16 source): ROWS x 32 bf16 tile, 256 thr,
// 4 threads/row, 16B per thread per chunk, tracked by ASYNCcnt.
// GVS addressing: uniform SGPR base + per-lane 32-bit byte offset.
// ---------------------------------------------------------------------------
template <int ROWS>
DEVINL void async_tile_bf16(const __bf16* __restrict__ src, int ld,
                            unsigned lds_base, int tid) {
  constexpr int CH = ROWS / 64;
  const int row0 = tid >> 2;
  const int col  = (tid & 3) * 8;
#pragma unroll
  for (int i = 0; i < CH; ++i) {
    const int row = row0 + i * 64;
    const unsigned lds  = lds_base + (unsigned)((row * LSTR + col) * 2);
    const unsigned goff = (unsigned)(row * ld + col) * 2u;
    asm volatile("global_load_async_to_lds_b128 %0, %1, %2"
                 :: "v"(lds), "v"(goff), "s"(src) : "memory");
  }
}

// Synchronous fp32 -> bf16 converting tile loader (for fp32 A operand).
template <int ROWS>
DEVINL void load_tile_f32(const float* __restrict__ src, int ld,
                          __bf16* dst, int tid) {
  constexpr int CH = ROWS / 64;
  const int row0 = tid >> 2;
  const int col  = (tid & 3) * 8;
#pragma unroll
  for (int i = 0; i < CH; ++i) {
    const int row = row0 + i * 64;
    const f32x4* p = (const f32x4*)(src + (long)row * ld + col);
    f32x4 f0 = p[0], f1 = p[1];
    union { __bf16 t[8]; u32x4 u; } pk;
    pk.t[0] = (__bf16)f0.x; pk.t[1] = (__bf16)f0.y;
    pk.t[2] = (__bf16)f0.z; pk.t[3] = (__bf16)f0.w;
    pk.t[4] = (__bf16)f1.x; pk.t[5] = (__bf16)f1.y;
    pk.t[6] = (__bf16)f1.z; pk.t[7] = (__bf16)f1.w;
    *(u32x4*)(dst + row * LSTR + col) = pk.u;
  }
}

DEVINL void wait_async_all() {
  asm volatile("s_wait_asynccnt 0x0" ::: "memory");
}

// ---------------------------------------------------------------------------
// Generic WMMA GEMM: C[M,Nc] = A[M,K] * B[Nc,K]^T (+bias), A fp32-or-bf16,
// B bf16 K-major rows, C fp32 or bf16. Batched over blockIdx.z.
// C batch base = (z/hdiv)*cZ1 + (z%hdiv)*cZ2 (lets attn*V scatter per head).
// Block 256 = 8 waves; wave tile = WM*16 x WN*16; double-buffered LDS with
// async global->LDS copies overlapped with WMMA on the other buffer.
// ---------------------------------------------------------------------------
template <int BM, int BN, int WROWS, int WCOLS, int WM, int WN,
          typename AT, typename OT, bool BIAS>
__global__ __launch_bounds__(256)
void gemm_bf16_wmma(const AT* __restrict__ A, const __bf16* __restrict__ B,
                    const float* __restrict__ bias, OT* __restrict__ C,
                    int K, int lda, int ldb, int ldc,
                    long aZ, long bZ, long cZ1, long cZ2, int hdiv) {
  static_assert(BM == WROWS * WM * 16 && BN == WCOLS * WN * 16, "tile mismatch");
  __shared__ __bf16 As[2][BM * LSTR];
  __shared__ __bf16 Bs[2][BN * LSTR];

  const int tid  = threadIdx.x;
  const int wave = tid >> 5, lane = tid & 31;
  const int wr = wave / WCOLS, wc = wave % WCOLS;
  const int lr = lane & 15,  hi = lane >> 4;

  const int z = blockIdx.z;
  A += (long)z * aZ;
  B += (long)z * bZ;
  C += (long)(z / hdiv) * cZ1 + (long)(z % hdiv) * cZ2;

  const long tileM = (long)blockIdx.y * BM;
  const long tileN = (long)blockIdx.x * BN;
  const AT*     At = A + tileM * lda;
  const __bf16* Bt = B + tileN * ldb;

  // Workgroup-relative LDS byte offsets (flat-address low 32 bits).
  const unsigned asB[2] = {(unsigned)(unsigned long long)&As[0][0],
                           (unsigned)(unsigned long long)&As[1][0]};
  const unsigned bsB[2] = {(unsigned)(unsigned long long)&Bs[0][0],
                           (unsigned)(unsigned long long)&Bs[1][0]};

  const f32x8 zero8 = {0.f, 0.f, 0.f, 0.f, 0.f, 0.f, 0.f, 0.f};
  f32x8 acc[WM][WN];
#pragma unroll
  for (int m = 0; m < WM; ++m)
#pragma unroll
    for (int n = 0; n < WN; ++n) acc[m][n] = zero8;

  auto loadA = [&](int kk, int buf) {
    if constexpr (sizeof(AT) == 2)
      async_tile_bf16<BM>((const __bf16*)At + (long)kk * KT, lda, asB[buf], tid);
    else
      load_tile_f32<BM>((const float*)At + (long)kk * KT, lda, As[buf], tid);
  };
  auto loadB = [&](int kk, int buf) {
    async_tile_bf16<BN>(Bt + (long)kk * KT, ldb, bsB[buf], tid);
  };

  loadA(0, 0);
  loadB(0, 0);
  wait_async_all();
  __syncthreads();

  const int nk = K / KT;
  for (int kk = 0; kk < nk; ++kk) {
    const int cur = kk & 1;
    if (kk + 1 < nk) {
      loadA(kk + 1, cur ^ 1);   // async copies overlap WMMA below
      loadB(kk + 1, cur ^ 1);
    }
    AFrag afr[WM], bfr[WN];
#pragma unroll
    for (int m = 0; m < WM; ++m) {
      // A 16x32 bf16 layout: lanes 0-15 hold K[0..7],K[16..23];
      // lanes 16-31 hold K[8..15],K[24..31] for row M = lane%16.
      const __bf16* p = &As[cur][(wr * WM * 16 + m * 16 + lr) * LSTR];
      afr[m].u[0] = *(const u32x4*)(p + hi * 8);
      afr[m].u[1] = *(const u32x4*)(p + 16 + hi * 8);
    }
#pragma unroll
    for (int n = 0; n < WN; ++n) {
      // B 32x16 bf16 layout: lanes 0-15 hold K[0..7],K[16..23], lanes 16-31 K[8..15],K[24..31]
      // for column N = lane%16 (== row of [Nc,K] operand).
      const __bf16* p = &Bs[cur][(wc * WN * 16 + n * 16 + lr) * LSTR];
      bfr[n].u[0] = *(const u32x4*)(p + hi * 8);
      bfr[n].u[1] = *(const u32x4*)(p + 16 + hi * 8);
    }
#pragma unroll
    for (int m = 0; m < WM; ++m)
#pragma unroll
      for (int n = 0; n < WN; ++n)
        acc[m][n] = __builtin_amdgcn_wmma_f32_16x16x32_bf16(
            false, afr[m].v, false, bfr[n].v, (short)0, acc[m][n],
            false, false);
    wait_async_all();           // drain this iteration's async copies
    __syncthreads();
  }

  // D layout: VGPR v -> row (m*16 + hi*8 + v), lane lr -> col.
#pragma unroll
  for (int n = 0; n < WN; ++n) {
    const long col = tileN + wc * WN * 16 + n * 16 + lr;
    const float bv = BIAS ? bias[col] : 0.0f;
#pragma unroll
    for (int m = 0; m < WM; ++m) {
      const long row0 = tileM + wr * WM * 16 + m * 16 + hi * 8;
#pragma unroll
      for (int v = 0; v < 8; ++v) {
        const float val = acc[m][n][v] + bv;
        C[(row0 + v) * ldc + col] = (OT)val;
      }
    }
  }
}

// ---------------------------------------------------------------------------
// fp32 -> bf16 convert, 4 elements / thread
// ---------------------------------------------------------------------------
__global__ __launch_bounds__(256)
void cvt_kernel(const float* __restrict__ in, __bf16* __restrict__ out,
                long n4) {
  const long i = (long)blockIdx.x * 256 + threadIdx.x;
  if (i >= n4) return;
  f32x4 f = ((const f32x4*)in)[i];
  union { __bf16 t[4]; unsigned long long u; } pk;
  pk.t[0] = (__bf16)f.x; pk.t[1] = (__bf16)f.y;
  pk.t[2] = (__bf16)f.z; pk.t[3] = (__bf16)f.w;
  ((unsigned long long*)out)[i] = pk.u;
}

// ---------------------------------------------------------------------------
// Per-head-row LayerNorm of q and k (D=64, one wave32 per row, 2 elems/lane),
// q additionally scaled by D^-0.5; writes bf16 [B,H,N,D].
// ---------------------------------------------------------------------------
DEVINL void ln_row(const float* __restrict__ in, const float* __restrict__ g,
                   const float* __restrict__ be, __bf16* __restrict__ out,
                   int lane, float mul) {
  f32x2 x = *(const f32x2*)(in + lane * 2);
  float s = x.x + x.y;
  float q = x.x * x.x + x.y * x.y;
#pragma unroll
  for (int off = 16; off > 0; off >>= 1) {
    s += __shfl_xor(s, off);
    q += __shfl_xor(q, off);
  }
  const float mean = s * (1.0f / 64.0f);
  const float var  = q * (1.0f / 64.0f) - mean * mean;
  const float inv  = rsqrtf(var + 1e-5f) * mul;
  const float y0 = (x.x - mean) * inv * g[lane * 2 + 0] + be[lane * 2 + 0] * mul;
  const float y1 = (x.y - mean) * inv * g[lane * 2 + 1] + be[lane * 2 + 1] * mul;
  union { __bf16 t[2]; unsigned int u; } pk;
  pk.t[0] = (__bf16)y0; pk.t[1] = (__bf16)y1;
  *(unsigned int*)(out + lane * 2) = pk.u;
}

__global__ __launch_bounds__(256)
void ln_qk_kernel(const float* __restrict__ qkv,
                  const float* __restrict__ qg, const float* __restrict__ qb,
                  const float* __restrict__ kg, const float* __restrict__ kb,
                  __bf16* __restrict__ qn, __bf16* __restrict__ kn) {
  const int wave = threadIdx.x >> 5, lane = threadIdx.x & 31;
  const long r  = (long)blockIdx.x * 8 + wave;   // (b,h,n) row, r < B*H*N
  const int b   = (int)(r / (cH * cN));
  const int rem = (int)(r % (cH * cN));
  const int h   = rem / cN;
  const int n   = rem % cN;
  const long base = ((long)b * cN + n) * (3 * cC) + (long)h * cD;
  ln_row(qkv + base,      qg, qb, qn + r * cD, lane, cSCALE);
  ln_row(qkv + base + cC, kg, kb, kn + r * cD, lane, 1.0f);
}

// ---------------------------------------------------------------------------
// v -> vT bf16: vT[b,h,d,n] = qkv[b,n, 2C + h*64 + d]
// ---------------------------------------------------------------------------
__global__ __launch_bounds__(256)
void v_transpose_kernel(const float* __restrict__ qkv,
                        __bf16* __restrict__ vT) {
  const long i = (long)blockIdx.x * 256 + threadIdx.x;  // over B*H*D*N
  const int n = (int)(i & (cN - 1));
  long t = i >> 10;
  const int d = (int)(t & (cD - 1));
  t >>= 6;
  const int h = (int)(t % cH);
  const int b = (int)(t / cH);
  const float v = qkv[((long)b * cN + n) * (3 * cC) + 2 * cC + h * cD + d];
  vT[i] = (__bf16)v;
}

// ---------------------------------------------------------------------------
// Row softmax in place: one block per row of 1024 fp32, 4 elems/thread.
// ---------------------------------------------------------------------------
__global__ __launch_bounds__(256)
void softmax_kernel(float* __restrict__ attn) {
  float* p = attn + (long)blockIdx.x * 1024;
  const int tid = threadIdx.x;
  const int wave = tid >> 5, lane = tid & 31;
  f32x4 x = ((const f32x4*)p)[tid];

  float m = fmaxf(fmaxf(x.x, x.y), fmaxf(x.z, x.w));
#pragma unroll
  for (int off = 16; off > 0; off >>= 1) m = fmaxf(m, __shfl_xor(m, off));
  __shared__ float smax[8], ssum[8];
  if (lane == 0) smax[wave] = m;
  __syncthreads();
  float gm = smax[0];
#pragma unroll
  for (int i = 1; i < 8; ++i) gm = fmaxf(gm, smax[i]);

  const float e0 = __expf(x.x - gm), e1 = __expf(x.y - gm);
  const float e2 = __expf(x.z - gm), e3 = __expf(x.w - gm);
  float s = e0 + e1 + e2 + e3;
#pragma unroll
  for (int off = 16; off > 0; off >>= 1) s += __shfl_xor(s, off);
  if (lane == 0) ssum[wave] = s;
  __syncthreads();
  float gs = ssum[0];
#pragma unroll
  for (int i = 1; i < 8; ++i) gs += ssum[i];
  const float inv = 1.0f / gs;
  f32x4 r = {e0 * inv, e1 * inv, e2 * inv, e3 * inv};
  ((f32x4*)p)[tid] = r;
}

// ---------------------------------------------------------------------------
extern "C" void kernel_launch(void* const* d_in, const int* in_sizes, int n_in,
                              void* d_out, int out_size, void* d_ws,
                              size_t ws_size, hipStream_t stream) {
  const float* x      = (const float*)d_in[0];
  const float* qkv_w  = (const float*)d_in[1];
  const float* qkv_b  = (const float*)d_in[2];
  const float* q_g    = (const float*)d_in[3];
  const float* q_be   = (const float*)d_in[4];
  const float* k_g    = (const float*)d_in[5];
  const float* k_be   = (const float*)d_in[6];
  const float* proj_w = (const float*)d_in[7];
  const float* proj_b = (const float*)d_in[8];

  float* out  = (float*)d_out;                 // [B,N,C]
  float* attn = out + (long)cB * cN * cC;      // [B,H,N,N]

  char* ws = (char*)d_ws;
  __bf16* xb    = (__bf16*)ws; ws += (long)cB * cN * cC * 2;       // x bf16
  __bf16* wqkv  = (__bf16*)ws; ws += (long)3 * cC * cC * 2;        // qkv_w bf16
  __bf16* wproj = (__bf16*)ws; ws += (long)cC * cC * 2;            // proj_w bf16
  float*  qkvws = (float*)ws;  ws += (long)cB * cN * 3 * cC * 4;   // qkv fp32
  __bf16* qn    = (__bf16*)ws; ws += (long)cB * cH * cN * cD * 2;  // LN(q)*scale
  __bf16* kn    = (__bf16*)ws; ws += (long)cB * cH * cN * cD * 2;  // LN(k)
  __bf16* vT    = (__bf16*)ws; ws += (long)cB * cH * cD * cN * 2;  // v^T
  __bf16* aout  = (__bf16*)ws;                                     // [B,N,C]

  const dim3 blk(256);

  // 1) fp32 -> bf16 conversions (all counts divisible by 4)
  cvt_kernel<<<6144, blk, 0, stream>>>(x,      xb,    1572864);
  cvt_kernel<<<1728, blk, 0, stream>>>(qkv_w,  wqkv,  442368);
  cvt_kernel<<<576,  blk, 0, stream>>>(proj_w, wproj, 147456);

  // 2) QKV GEMM: [8192,768] x [2304,768]^T + bias -> fp32 [8192,2304]
  gemm_bf16_wmma<128, 128, 2, 4, 4, 2, __bf16, float, true>
      <<<dim3(2304 / 128, 8192 / 128, 1), blk, 0, stream>>>(
          xb, wqkv, qkv_b, qkvws, 768, 768, 768, 2304, 0, 0, 0, 0, 1);

  // 3) LayerNorm(q)*scale, LayerNorm(k) -> bf16 [B,H,N,D]
  ln_qk_kernel<<<12288, blk, 0, stream>>>(qkvws, q_g, q_be, k_g, k_be, qn, kn);

  // 4) v -> vT bf16 [B,H,D,N]
  v_transpose_kernel<<<24576, blk, 0, stream>>>(qkvws, vT);

  // 5) S = q @ k^T per head: 96 x [1024,64]x[1024,64]^T -> fp32 attn
  gemm_bf16_wmma<128, 128, 2, 4, 4, 2, __bf16, float, false>
      <<<dim3(8, 8, 96), blk, 0, stream>>>(
          qn, kn, nullptr, attn, 64, 64, 64, 1024,
          (long)cN * cD, (long)cN * cD, (long)cN * cN, 0, 1);

  // 6) softmax rows in place (B*H*N rows)
  softmax_kernel<<<98304, blk, 0, stream>>>(attn);

  // 7) out_h = attn @ v per head: [1024,1024] x [64,1024]^T -> bf16,
  //    scattered into [B,N,C] via per-head base (z/12)*N*C + (z%12)*64
  gemm_bf16_wmma<128, 64, 4, 2, 2, 2, float, __bf16, false>
      <<<dim3(1, 8, 96), blk, 0, stream>>>(
          attn, vT, nullptr, aout, 1024, 1024, 1024, 768,
          (long)cN * cN, (long)cD * cN, (long)cN * cC, cD, cH);

  // 8) proj GEMM: [8192,768] x [768,768]^T + bias -> fp32 out
  gemm_bf16_wmma<128, 128, 2, 4, 4, 2, __bf16, float, true>
      <<<dim3(768 / 128, 8192 / 128, 1), blk, 0, stream>>>(
          aout, wproj, proj_b, out, 768, 768, 768, 768, 0, 0, 0, 0, 1);

  (void)in_sizes; (void)n_in; (void)out_size; (void)ws_size;
}
